// NeuralSemiLagrangian_66090956751362
// MI455X (gfx1250) — matrix operand order repaired and checked by reading, not compiled, not run on hardware
//
#include <hip/hip_runtime.h>
#include <hip/hip_bf16.h>
#include <math.h>

#define H     721
#define W     1440
#define HW    (H*W)
#define CIN   40
#define KTOT  360          // 9 taps * 40 ic
#define KSTEPS 12
#define WSTR  392          // wpad row stride (784B, bank-conflict-free b128)
#define ZROW  (CIN*3*100)  // zeroed row in raw[] for K-tail padding
#define TILE_A 96          // pixels per WG in conv1 (15 * 96 = 1440)
#define NXA   (W/TILE_A)   // 15
#define NWGA  (NXA*H)      // 10815
#define TILE_D 240         // pixels per WG in advect (6 * 240 = 1440)
#define NXD   (W/TILE_D)   // 6
#define DSTR  242          // advect tile row stride (col 0 and 241 are margins)

typedef __attribute__((ext_vector_type(16))) __bf16 v16bf;
typedef __attribute__((ext_vector_type(8)))  float  v8f;
typedef __attribute__((ext_vector_type(4)))  unsigned int v4u;
typedef __attribute__((ext_vector_type(8)))  int    v8i;
typedef __attribute__((ext_vector_type(4)))  int    v4i;

static __device__ __forceinline__ unsigned short f2bf(float f) {
  unsigned int x = __float_as_uint(f);
  x += 0x7fffu + ((x >> 16) & 1u);           // round to nearest even
  return (unsigned short)(x >> 16);
}
static __device__ __forceinline__ float bf2f(unsigned short u) {
  return __uint_as_float(((unsigned int)u) << 16);
}

// Map geo_cyclic_pad(PAD=1) coordinates (py in [0,H+1], px in [0,W+1])
// back to a flat offset sy*W+sx into the unpadded (H,W) plane.
static __device__ __forceinline__ int geo_map(int py, int px) {
  int mx = (px - 1 + W) % W;                 // lon wrap
  int sy, sx;
  if (py == 0)        { sy = 0;     sx = (mx + W/2) % W; }   // pole flip+roll
  else if (py == H+1) { sy = H - 1; sx = (mx + W/2) % W; }
  else                { sy = py - 1; sx = mx; }
  return sy * W + sx;
}

// raw[] offset for K index k = tap*40+ic (compile-time after unrolling)
static __device__ __forceinline__ constexpr int aoff(int k) {
  int tap = k / CIN, ic = k - tap * CIN;
  int ky = tap / 3,  kx = tap - ky * 3;
  return (ic * 3 + ky) * 100 + kx;
}

static __device__ __forceinline__ float cubicw(float xx) {
  const float A = -0.75f;
  float ax = fabsf(xx);
  float ax2 = ax * ax, ax3 = ax2 * ax;
  float w1 = (A + 2.f) * ax3 - (A + 3.f) * ax2 + 1.f;
  float w2 = A * ax3 - 5.f * A * ax2 + 8.f * A * ax - 4.f * A;
  return ax <= 1.f ? w1 : (ax < 2.f ? w2 : 0.f);
}

// CDNA5 async global->LDS copy (ASYNCcnt-tracked), 16B granule.
static __device__ __forceinline__ void async_b128(unsigned lds_off, const void* src) {
  asm volatile("global_load_async_to_lds_b128 %0, %1, off"
               :: "v"(lds_off), "v"((unsigned long long)(size_t)src)
               : "memory");
}
static __device__ __forceinline__ void wait_async0() {
  asm volatile("s_wait_asynccnt 0x0" ::: "memory");
}

// CDNA5 Tensor Data Mover: one-row (1D) tile load, bf16 elements.
// D# packing per CDNA5 ISA (groups 2/3 zero = <=2D).
// This toolchain's builtin is the 6-arg clang-23 form:
//   (uint32x4 g0, int32x8 g1, int32x4, int32x4, int32x8, i32 cpol)
static __device__ __forceinline__ void tdm_load_row(unsigned lds_off, const void* gsrc,
                                                    int nelem) {
  unsigned long long ga = (unsigned long long)(size_t)gsrc;
  v4u g0;
  g0[0] = 1u;                                        // count=1 (valid), user mode
  g0[1] = lds_off;                                   // lds_addr [63:32]
  g0[2] = (unsigned)(ga & 0xffffffffu);              // global_addr [95:64]
  g0[3] = (unsigned)((ga >> 32) & 0x01ffffffu)       // global_addr [120:96]
        | (2u << 30);                                // type=2 ("image") [127:126]
  v8i g1;
  g1[0] = 0x10000;                                   // data_size=1 (2B) [17:16]
  g1[1] = (int)(((unsigned)nelem & 0xffffu) << 16);  // tensor_dim0[15:0]  @ [63:48]
  g1[2] = (int)(((unsigned)nelem >> 16)              // tensor_dim0[31:16] @ [79:64]
        | (1u << 16));                               // tensor_dim1=1      @ [111:80]
  g1[3] = (int)(((unsigned)nelem & 0xffffu) << 16);  // tile_dim0          @ [127:112]
  g1[4] = 1;                                         // tile_dim1=1, tile_dim2=0
  g1[5] = nelem;                                     // tensor_dim0_stride (lo)
  g1[6] = 0;
  g1[7] = 0;
  v4i z4 = {0, 0, 0, 0};
  v8i z8 = {0, 0, 0, 0, 0, 0, 0, 0};
  __builtin_amdgcn_tensor_load_to_lds(g0, g1, z4, z4, z8, 0);
}

// ---------------------------------------------------------------------------
// Kernel A: conv1 (3x3, 40->40) as implicit GEMM on bf16 WMMA.
//   M = 16 pixels/subtile, N = 16 output channels/octile (3 octiles -> 48),
//   K = tap*40 + ic, padded 360 -> 384, 12 steps of v_wmma_f32_16x16x32_bf16.
// Also emits per-WG sum / sumsq partials for the global LayerNorm.
// ---------------------------------------------------------------------------
__global__ __launch_bounds__(192)
void conv1_wmma_kernel(const float* __restrict__ dyn, const float* __restrict__ sta,
                       const float* __restrict__ w1,  const float* __restrict__ b1,
                       unsigned short* __restrict__ h_ws, float* __restrict__ partials)
{
  __shared__ unsigned short raw[CIN*3*100 + 100];  // + zeroed row at ZROW
  __shared__ unsigned short wpad[48*WSTR];         // zero-padded weights (bf16)
  __shared__ float redS[8], redS2[8];

  const int tid = threadIdx.x;
  const int y0  = blockIdx.y;                 // output row
  const int x0  = blockIdx.x * TILE_A;        // first output column

  // ---- stage weights: wpad[n][k], k = tap*40+ic -------------------------
  for (int idx = tid; idx < 48*WSTR; idx += 192) {
    int n = idx / WSTR, k = idx - n*WSTR;
    unsigned short v = 0;
    if (n < CIN && k < KTOT) {
      int tap = k / CIN, ic = k - tap*CIN;
      int ky = tap / 3,  kx = tap - ky*3;
      v = f2bf(w1[((n*CIN + ic)*3 + ky)*3 + kx]);
    }
    wpad[idx] = v;
  }
  // ---- stage input: 40ch x 3 padded rows (y0..y0+2) x 98 cols -----------
  for (int idx = tid; idx < CIN*3*98; idx += 192) {
    int ic  = idx / (3*98);
    int rem = idx - ic*3*98;
    int r   = rem / 98, col = rem - r*98;
    int off = geo_map(y0 + r, x0 + col);
    float v = (ic < 32) ? dyn[ic*HW + off] : sta[(ic - 32)*HW + off];
    raw[(ic*3 + r)*100 + col] = f2bf(v);
  }
  if (tid < 100) raw[ZROW + tid] = 0;         // zero row for K-tail
  __syncthreads();

  const int wave = tid >> 5;                  // subtile 0..5 (16 pixels each)
  const int lane = tid & 31;
  const int grp  = lane >> 4;
  const int m    = lane & 15;
  const int colbase = wave*16 + m;
  const int colG    = colbase + grp*2400;     // hoisted: +8 ic == +2400 elements

  v8f c[3] = {v8f{}, v8f{}, v8f{}};
  union Frag { v16bf v; unsigned short u[16]; };

  #pragma unroll
  for (int ks = 0; ks < KSTEPS; ++ks) {
    const int kbase = ks * 32;
    // A fragment: most elements use the hoisted base + constant immediate;
    // tap-crossing elements need one cndmask; the K>=360 tail is literal zero.
    Frag a;
    #pragma unroll
    for (int e = 0; e < 16; ++e) {
      const int k0 = kbase + e + ((e >= 8) ? 8 : 0);
      const int k1 = k0 + 8;
      if (k1 < KTOT && aoff(k1) == aoff(k0) + 2400) {
        a.u[e] = raw[colG + aoff(k0)];                       // imm-offset ds_load_u16
      } else if (k1 < KTOT) {
        a.u[e] = raw[colbase + (grp ? aoff(k1) : aoff(k0))]; // tap crossing
      } else if (k0 < KTOT) {
        a.u[e] = raw[colbase + (grp ? ZROW : aoff(k0))];     // half-valid tail
      } else {
        a.u[e] = 0;                                          // dead tail
      }
    }
    #pragma unroll
    for (int t = 0; t < 3; ++t) {
      const int n = t*16 + m;                 // output channel column
      Frag b;                                 // 2 runs of 8 bf16 (k=8h.., 16+8h..) -> 2x ds_load_b128
      #pragma unroll
      for (int e = 0; e < 16; ++e)
        b.u[e] = wpad[n*WSTR + kbase + e + ((e >= 8) ? 8 : 0) + (grp ? 8 : 0)];
      c[t] = __builtin_amdgcn_wmma_f32_16x16x32_bf16(false, a.v, false, b.v,
                                                     (short)0, c[t], false, false);
    }
  }

  // ---- epilogue: bias, LN partial sums, transpose, store ----------------
  float s = 0.f, s2 = 0.f;
  float vals[3][8];
  #pragma unroll
  for (int t = 0; t < 3; ++t) {
    int oc = t*16 + m;
    float bv = (oc < CIN) ? b1[oc] : 0.f;
    #pragma unroll
    for (int r = 0; r < 8; ++r) {
      float v = c[t][r] + bv;
      vals[t][r] = v;
      if (oc < CIN) { s += v; s2 += v*v; }
    }
  }
  #pragma unroll
  for (int o = 16; o > 0; o >>= 1) { s += __shfl_xor(s, o); s2 += __shfl_xor(s2, o); }
  if (lane == 0) { redS[wave] = s; redS2[wave] = s2; }
  __syncthreads();
  if (tid == 0) {
    float ts = 0.f, ts2 = 0.f;
    for (int i = 0; i < 6; ++i) { ts += redS[i]; ts2 += redS2[i]; }
    int wg = blockIdx.y * gridDim.x + blockIdx.x;
    partials[2*wg] = ts; partials[2*wg + 1] = ts2;
  }
  // transpose through LDS (reuse raw) so the global store is oc-row coalesced
  unsigned short* tbuf = raw;                 // needs 48*100 <= 12000 entries
  __syncthreads();
  #pragma unroll
  for (int t = 0; t < 3; ++t) {
    int oc = t*16 + m;
    #pragma unroll
    for (int r = 0; r < 8; ++r) {
      int px = wave*16 + r + 8*grp;           // C-matrix M mapping
      tbuf[oc*100 + px] = f2bf(vals[t][r]);
    }
  }
  __syncthreads();
  for (int idx = tid; idx < CIN*TILE_A; idx += 192) {
    int oc = idx / TILE_A, col = idx - oc*TILE_A;
    h_ws[(oc*H + y0)*W + x0 + col] = tbuf[oc*100 + col];
  }
}

// ---------------------------------------------------------------------------
// Kernel B: deterministic reduction of per-WG partials -> mu, rsqrt(var+eps)
// ---------------------------------------------------------------------------
__global__ __launch_bounds__(256)
void stats_kernel(const float* __restrict__ partials, float* __restrict__ stats)
{
  __shared__ float rsS[8], rsS2[8];
  float s = 0.f, s2 = 0.f;
  for (int i = threadIdx.x; i < NWGA; i += 256) { s += partials[2*i]; s2 += partials[2*i+1]; }
  #pragma unroll
  for (int o = 16; o > 0; o >>= 1) { s += __shfl_xor(s, o); s2 += __shfl_xor(s2, o); }
  int wave = threadIdx.x >> 5, lane = threadIdx.x & 31;
  if (lane == 0) { rsS[wave] = s; rsS2[wave] = s2; }
  __syncthreads();
  if (threadIdx.x == 0) {
    float ts = 0.f, ts2 = 0.f;
    for (int i = 0; i < 8; ++i) { ts += rsS[i]; ts2 += rsS2[i]; }
    const float N = (float)CIN * (float)HW;
    float mu  = ts / N;
    float var = ts2 / N - mu * mu;
    stats[0] = mu;
    stats[1] = rsqrtf(var + 1e-5f);
  }
}

// ---------------------------------------------------------------------------
// Kernel C: LayerNorm affine + SiLU, in place on bf16 h buffer.
// ln_w/ln_b (166 MB each) are streamed exactly once, fully coalesced.
// ---------------------------------------------------------------------------
__global__ __launch_bounds__(256)
void ln_silu_kernel(unsigned short* __restrict__ h_ws,
                    const float* __restrict__ lnw, const float* __restrict__ lnb,
                    const float* __restrict__ stats)
{
  int pix = blockIdx.x * 256 + threadIdx.x;
  if (pix >= HW) return;
  const float mu = stats[0], rs = stats[1];
  #pragma unroll 4
  for (int cc = 0; cc < CIN; ++cc) {
    int idx = cc*HW + pix;
    if (cc + 1 < CIN) __builtin_prefetch(&lnw[idx + HW], 0, 0);   // global_prefetch_b8
    float hv = bf2f(h_ws[idx]);
    float v  = (hv - mu) * rs * lnw[idx] + lnb[idx];
    float sg = 1.0f / (1.0f + __expf(-v));
    h_ws[idx] = f2bf(v * sg);
  }
}

// ---------------------------------------------------------------------------
// Kernel D: conv2 (3x3, 40->2) from a TDM-staged h_silu tile, spherical
// departure points, and bicubic-border resample through the pad mapping.
// Interior rows: 120 tensor_load_to_lds descriptors (TENSORcnt), one per
// (ic, ky) row-run, each placed at its own LDS row with margin columns.
// Pole blocks (y==0 / y==H-1): scalar staging. conv2 weights: async B128.
// ---------------------------------------------------------------------------
__global__ __launch_bounds__(256)
void advect_kernel(const unsigned short* __restrict__ hs,
                   const float* __restrict__ w2g, const float* __restrict__ b2,
                   const float* __restrict__ dyn, const int* __restrict__ dtp,
                   float* __restrict__ out)
{
  __shared__ __attribute__((aligned(16))) unsigned short tile[CIN*3*DSTR];
  __shared__ __attribute__((aligned(16))) float w2s[2*CIN*9];

  const int tid = threadIdx.x;
  const int y   = blockIdx.y;
  const int x0  = blockIdx.x * TILE_D;

  const unsigned tile_base = (unsigned)(size_t)(void*)tile;
  const unsigned w2s_base  = (unsigned)(size_t)(void*)w2s;

  // conv2 weights: 720 f32 = 180 aligned 16B chunks, async copied (ASYNCcnt)
  if (tid < 180) async_b128(w2s_base + tid*16, (const char*)w2g + tid*16);

  if (y >= 1 && y <= H-2) {
    // Interior: all 3 padded rows are regular strips -> Tensor Data Mover.
    if (tid == 0) {
      for (int run = 0; run < CIN*3; ++run) {
        int ic = run / 3, rr = run - ic*3;
        const unsigned short* src = hs + (size_t)ic*HW + (size_t)(y - 1 + rr)*W + x0;
        tdm_load_row(tile_base + (unsigned)(run*DSTR + 1)*2u, src, TILE_D);
      }
      __builtin_amdgcn_s_wait_tensorcnt(0);
    }
    // margin columns (col 0 and 241)
    if (tid < CIN*3*2) {
      int run = tid >> 1;
      int col = (tid & 1) ? (TILE_D + 1) : 0;
      int ic  = run / 3, rr = run - ic*3;
      int off = geo_map(y + rr, x0 + col);
      tile[run*DSTR + col] = hs[(size_t)ic*HW + off];
    }
  } else {
    // Pole blocks: scalar staging through the flip+roll mapping.
    for (int idx = tid; idx < CIN*3*DSTR; idx += 256) {
      int run = idx / DSTR, col = idx - run*DSTR;
      int ic  = run / 3,    rr  = run - ic*3;
      int off = geo_map(y + rr, x0 + col);
      tile[run*DSTR + col] = hs[(size_t)ic*HW + off];
    }
  }
  wait_async0();
  __syncthreads();

  if (tid >= TILE_D) return;
  const int x = x0 + tid;

  // conv2: two output channels; tile row holds padded cols 0..241 directly
  float u = b2[0], v = b2[1];
  #pragma unroll
  for (int ky = 0; ky < 3; ++ky)
    #pragma unroll
    for (int kx = 0; kx < 3; ++kx)
      for (int ic = 0; ic < CIN; ++ic) {
        float hv = bf2f(tile[(ic*3 + ky)*DSTR + tid + kx]);
        u += hv * w2s[(ic*9) + ky*3 + kx];
        v += hv * w2s[CIN*9 + (ic*9) + ky*3 + kx];
      }

  // departure points on the sphere
  const float PI = 3.14159265358979323846f;
  const float TWO_PI = 2.f * PI;
  const float fdt = (float)(*dtp);
  float lat_g = -0.5f*PI + PI * ((float)y / (float)(H - 1));
  float lon_g = TWO_PI * ((float)x / (float)(W - 1));
  float lonp = -u * fdt, latp = -v * fdt;
  float slp, clp, slo, clo, slg, clg;
  __sincosf(latp, &slp, &clp);
  __sincosf(lonp, &slo, &clo);
  __sincosf(lat_g, &slg, &clg);

  float sin_lat = slp*clg + clp*clo*slg;
  sin_lat = fminf(fmaxf(sin_lat, -1.f + 1e-7f), 1.f - 1e-7f);
  float lat_dep = asinf(sin_lat);
  float num = clp * slo;
  float den = clp*clo*clg - slp*slg;
  float lon_dep = lon_g + atan2f(num, den);
  if (lon_dep < 0.f)      lon_dep += TWO_PI;
  if (lon_dep > TWO_PI)   lon_dep -= TWO_PI;

  const int Hp = H + 2, Wp = W + 2;
  float gx = (lon_dep / TWO_PI * 2.f - 1.f) * ((float)W / (float)Wp);
  float gy = (lat_dep / PI     * 2.f - 1.f) * ((float)H / (float)Hp);
  float ix = fminf(fmaxf((gx + 1.f) * 0.5f * (float)(Wp - 1), 0.f), (float)(Wp - 1));
  float iy = fminf(fmaxf((gy + 1.f) * 0.5f * (float)(Hp - 1), 0.f), (float)(Hp - 1));
  float xf = floorf(ix), yf = floorf(iy);
  float tx = ix - xf,   ty = iy - yf;
  int x0i = (int)xf,    y0i = (int)yf;

  float wxv[4] = { cubicw(tx + 1.f), cubicw(tx), cubicw(1.f - tx), cubicw(2.f - tx) };
  float wyv[4] = { cubicw(ty + 1.f), cubicw(ty), cubicw(1.f - ty), cubicw(2.f - ty) };

  int offs[16];
  float wc[16];
  #pragma unroll
  for (int i = 0; i < 4; ++i) {
    int yi = min(max(y0i + i - 1, 0), Hp - 1);
    #pragma unroll
    for (int j = 0; j < 4; ++j) {
      int xi = min(max(x0i + j - 1, 0), Wp - 1);
      offs[i*4 + j] = geo_map(yi, xi);       // dyn_pad never materialized
      wc[i*4 + j]   = wyv[i] * wxv[j];
    }
  }

  const int opix = y*W + x;
  for (int cc = 0; cc < 32; ++cc) {
    const float* base = dyn + cc*HW;
    float acc = 0.f;
    #pragma unroll
    for (int q = 0; q < 16; ++q) acc += wc[q] * base[offs[q]];
    out[cc*HW + opix] = acc;
  }
}

// ---------------------------------------------------------------------------
extern "C" void kernel_launch(void* const* d_in, const int* in_sizes, int n_in,
                              void* d_out, int out_size, void* d_ws, size_t ws_size,
                              hipStream_t stream)
{
  const float* dyn = (const float*)d_in[0];
  const float* sta = (const float*)d_in[1];
  const float* w1  = (const float*)d_in[2];
  const float* b1  = (const float*)d_in[3];
  const float* lnw = (const float*)d_in[4];
  const float* lnb = (const float*)d_in[5];
  const float* w2  = (const float*)d_in[6];
  const float* b2  = (const float*)d_in[7];
  const int*   dt  = (const int*)d_in[8];
  float* out = (float*)d_out;

  unsigned short* h_ws = (unsigned short*)d_ws;                      // 40*HW bf16
  float* partials = (float*)((char*)d_ws + (size_t)CIN*HW*2);        // 2*NWGA f32
  float* stats    = partials + 2*NWGA;                               // mu, rsigma

  conv1_wmma_kernel<<<dim3(NXA, H), 192, 0, stream>>>(dyn, sta, w1, b1, h_ws, partials);
  stats_kernel<<<1, 256, 0, stream>>>(partials, stats);
  ln_silu_kernel<<<(HW + 255)/256, 256, 0, stream>>>(h_ws, lnw, lnb, stats);
  advect_kernel<<<dim3(NXD, H), 256, 0, stream>>>(h_ws, w2, b2, dyn, dt, out);
}
